// GCN2_24257975288510
// MI455X (gfx1250) — matrix operand order repaired and checked by reading, not compiled, not run on hardware
//
#include <hip/hip_runtime.h>
#include <hip/hip_bf16.h>
#include <math.h>

#define NNODES 20000
#define NEDGES 320000
#define DIN    128
#define DH     304
#define KPAD   320
#define RALL   (3 * NNODES)   // 60000 rows across the 3 graphs

typedef __attribute__((ext_vector_type(16))) __bf16       v16bf;
typedef __attribute__((ext_vector_type(8)))  float        v8f;
typedef __attribute__((ext_vector_type(8)))  unsigned int v8u;

__device__ __forceinline__ unsigned short f2bf(float f) {
    unsigned int u = __float_as_uint(f);
    unsigned int r = u + 0x7FFFu + ((u >> 16) & 1u);   // round-to-nearest-even
    return (unsigned short)(r >> 16);
}

// ---------------------------------------------------------------- utilities
__global__ void zero_f32(float* __restrict__ p, int n) {
    int i = blockIdx.x * blockDim.x + threadIdx.x;
    if (i < n) p[i] = 0.0f;
}

// ------------------------------------------------------------- degree prep
__global__ void deg_count(float* __restrict__ dego, float* __restrict__ degi,
                          const int* __restrict__ src, const int* __restrict__ dst,
                          int E) {
    int e = blockIdx.x * blockDim.x + threadIdx.x;
    if (e >= E) return;
    atomicAdd(&dego[src[e]], 1.0f);
    atomicAdd(&degi[dst[e]], 1.0f);
}

__global__ void finish_deg(float* __restrict__ rso, float* __restrict__ rsi, int n) {
    int i = blockIdx.x * blockDim.x + threadIdx.x;
    if (i >= n) return;
    rso[i] = rsqrtf(fmaxf(rso[i], 1.0f));
    rsi[i] = rsqrtf(fmaxf(rsi[i], 1.0f));
}

// ------------------------------------------------- weight pack: W[KxM] -> Wt bf16 [M x Ks]
__global__ void pack_wt(const float* __restrict__ W, unsigned short* __restrict__ wt,
                        int K, int Ks, int M) {
    int k = blockIdx.x * blockDim.x + threadIdx.x;
    int n = blockIdx.y;
    if (k >= Ks) return;
    wt[(size_t)n * Ks + k] = (k < K) ? f2bf(W[(size_t)k * M + n]) : (unsigned short)0;
}

// ------------------------------------------------- SpMM: agg[dst] += feat[src]*rso[src]
__global__ void spmm_scatter(const float* __restrict__ feat, int fstride,
                             float* __restrict__ agg, int astride,
                             const float* __restrict__ rso,
                             const int* __restrict__ src, const int* __restrict__ dst,
                             int D) {
    int e = blockIdx.x;
    int f = blockIdx.y * blockDim.x + threadIdx.x;
    if (f >= D) return;
    int s = src[e], d = dst[e];
    float v = feat[(size_t)s * fstride + f] * rso[s];
    atomicAdd(&agg[(size_t)d * astride + f], v);
}

// ------------------------------------------------- scale by rsqrt(in_deg), cvt to bf16
__global__ void scale_to_bf16(const float* __restrict__ agg, int astride,
                              const float* __restrict__ rsi,
                              unsigned short* __restrict__ hb, int Ks) {
    int r = blockIdx.y;
    int k = blockIdx.x * blockDim.x + threadIdx.x;
    if (k >= Ks) return;
    hb[(size_t)r * Ks + k] = f2bf(agg[(size_t)r * astride + k] * rsi[r]);
}

// ------------------------------------------------- WMMA GEMM: out = relu(A @ Wt^T + b)
// A: bf16 [R x Ks] row-major, Bt: bf16 [DH x Ks] (i.e. W transposed), out: f32 [R x DH]
__global__ void wmma_gemm_relu(const unsigned short* __restrict__ A,
                               const unsigned short* __restrict__ Bt,
                               const float* __restrict__ bias,
                               float* __restrict__ out,
                               int Ks, int n_tiles) {
    const int NT = DH / 16;                          // 19 column tiles, exact
    int wave = threadIdx.x >> 5;
    int lane = threadIdx.x & 31;
    int tile = blockIdx.x * (blockDim.x >> 5) + wave;
    if (tile >= n_tiles) return;
    int m0   = (tile / NT) * 16;
    int n0   = (tile % NT) * 16;
    int half = lane >> 4;                            // 0: lanes 0-15, 1: lanes 16-31
    int l15  = lane & 15;

    const unsigned int* ap = (const unsigned int*)(A  + (size_t)(m0 + l15) * Ks);
    const unsigned int* bp = (const unsigned int*)(Bt + (size_t)(n0 + l15) * Ks);
    float bsc = bias[n0 + l15];

    v8f acc = {};
    int kd = 0;                                      // dword offset (= k0/2)
    for (int k0 = 0; k0 < Ks; k0 += 32, kd += 16) {
        v8u au, bu;
        // A 16x32 bf16 layout: vgprs 0-3 -> K = 8*half + 2v ; vgprs 4-7 -> 16 + 8*half + 2(v-4)
        int ha = kd + half * 4;
        au[0] = ap[ha + 0];  au[1] = ap[ha + 1];
        au[2] = ap[ha + 2];  au[3] = ap[ha + 3];
        au[4] = ap[ha + 8];  au[5] = ap[ha + 9];
        au[6] = ap[ha + 10]; au[7] = ap[ha + 11];
        // B 32x16 bf16 layout: vgprs 0-3 -> K = 8*half + 2v ; vgprs 4-7 -> 16 + 8*half + 2(v-4), lane -> column
        int hb2 = kd + half * 4;
#pragma unroll
        for (int v = 0; v < 8; ++v) bu[v] = bp[hb2 + v + (v >= 4 ? 4 : 0)];

        v16bf a = __builtin_bit_cast(v16bf, au);
        v16bf b = __builtin_bit_cast(v16bf, bu);
        acc = __builtin_amdgcn_wmma_f32_16x16x32_bf16(
            false, a, false, b, (short)0, acc, false, false);
    }

    int col = n0 + l15;
#pragma unroll
    for (int r = 0; r < 8; ++r) {
        float v = acc[r] + bsc;
        out[(size_t)(m0 + r + 8 * half) * DH + col] = v > 0.0f ? v : 0.0f;
    }
}

// ------------------------------------------------- max pool over rows (values >= 0)
__global__ void pool_max(const float* __restrict__ h, float* __restrict__ pool, int R) {
    int n = blockIdx.x * blockDim.x + threadIdx.x;
    if (n >= DH) return;
    int r0 = blockIdx.y * 256;
    int r1 = r0 + 256; if (r1 > R) r1 = R;
    float m = 0.0f;
    for (int r = r0; r < r1; ++r) m = fmaxf(m, h[(size_t)r * DH + n]);
    atomicMax((int*)&pool[n], __float_as_int(m));   // valid: all values >= 0
}

// ------------------------------------------------- MLP head: 304 -> 128 -> 64 -> 1
__global__ void mlp_head(const float* __restrict__ pool,
                         const float* __restrict__ fW1, const float* __restrict__ fb1,
                         const float* __restrict__ fW2, const float* __restrict__ fb2,
                         const float* __restrict__ fW3, const float* __restrict__ fb3,
                         float* __restrict__ out) {
    __shared__ float sp[DH];
    __shared__ float s1[128];
    __shared__ float s2[64];
    int t = threadIdx.x;
    for (int i = t; i < DH; i += blockDim.x) sp[i] = pool[i];
    __syncthreads();
    if (t < 128) {
        float acc = fb1[t];
        for (int i = 0; i < DH; ++i) acc += sp[i] * fW1[(size_t)i * 128 + t];
        s1[t] = acc > 0.0f ? acc : 0.0f;
    }
    __syncthreads();
    if (t < 64) {
        float acc = fb2[t];
        for (int i = 0; i < 128; ++i) acc += s1[i] * fW2[(size_t)i * 64 + t];
        s2[t] = acc > 0.0f ? acc : 0.0f;
    }
    __syncthreads();
    if (t == 0) {
        float acc = fb3[0];
        for (int i = 0; i < 64; ++i) acc += s2[i] * fW3[i];
        out[0] = 1.0f / (1.0f + expf(-acc));
    }
}

// =========================================================================
extern "C" void kernel_launch(void* const* d_in, const int* in_sizes, int n_in,
                              void* d_out, int out_size, void* d_ws, size_t ws_size,
                              hipStream_t stream) {
    (void)in_sizes; (void)n_in; (void)out_size; (void)ws_size;

    const float* x[3]   = { (const float*)d_in[0], (const float*)d_in[1], (const float*)d_in[2] };
    const int*   src[3] = { (const int*)d_in[3], (const int*)d_in[5], (const int*)d_in[7] };
    const int*   dst[3] = { (const int*)d_in[4], (const int*)d_in[6], (const int*)d_in[8] };
    const float* W[3]   = { (const float*)d_in[9],  (const float*)d_in[11], (const float*)d_in[13] };
    const float* b[3]   = { (const float*)d_in[10], (const float*)d_in[12], (const float*)d_in[14] };
    const float* fW1 = (const float*)d_in[15]; const float* fb1 = (const float*)d_in[16];
    const float* fW2 = (const float*)d_in[17]; const float* fb2 = (const float*)d_in[18];
    const float* fW3 = (const float*)d_in[19]; const float* fb3 = (const float*)d_in[20];

    // ---- workspace carve-up (256B aligned) ----
    size_t off = 0;
    char* base = (char*)d_ws;
    auto carve = [&](size_t bytes) -> char* {
        char* p = base + off;
        off += (bytes + 255) & ~(size_t)255;
        return p;
    };
    float*          bufA = (float*)         carve((size_t)RALL * DH   * 4); // layer outputs
    float*          bufB = (float*)         carve((size_t)RALL * KPAD * 4); // aggregation
    unsigned short* hb   = (unsigned short*)carve((size_t)RALL * KPAD * 2); // bf16 activations
    unsigned short* wt1  = (unsigned short*)carve((size_t)DH * DIN  * 2);
    unsigned short* wt2  = (unsigned short*)carve((size_t)DH * KPAD * 2);
    unsigned short* wt3  = (unsigned short*)carve((size_t)DH * KPAD * 2);
    float*          rso  = (float*)         carve((size_t)RALL * 4);
    float*          rsi  = (float*)         carve((size_t)RALL * 4);
    float*          pool = (float*)         carve((size_t)DH * 4);

    // ---- degree normalization ----
    zero_f32<<<(RALL + 255) / 256, 256, 0, stream>>>(rso, RALL);
    zero_f32<<<(RALL + 255) / 256, 256, 0, stream>>>(rsi, RALL);
    for (int g = 0; g < 3; ++g)
        deg_count<<<(NEDGES + 255) / 256, 256, 0, stream>>>(rso + (size_t)g * NNODES,
                                                            rsi + (size_t)g * NNODES,
                                                            src[g], dst[g], NEDGES);
    finish_deg<<<(RALL + 255) / 256, 256, 0, stream>>>(rso, rsi, RALL);

    // ---- pack weights to transposed bf16 (K zero-padded to multiple of 32) ----
    pack_wt<<<dim3(1, DH), 256, 0, stream>>>(W[0], wt1, DIN, DIN,  DH);
    pack_wt<<<dim3(2, DH), 256, 0, stream>>>(W[1], wt2, DH,  KPAD, DH);
    pack_wt<<<dim3(2, DH), 256, 0, stream>>>(W[2], wt3, DH,  KPAD, DH);

    const unsigned short* wts[3] = { wt1, wt2, wt3 };
    const int Ks_l[3] = { DIN, KPAD, KPAD };
    const int D_l[3]  = { DIN, DH,   DH   };
    const int n_tiles = (RALL / 16) * (DH / 16);   // 3750 * 19 = 71250, exact tiling

    for (int l = 0; l < 3; ++l) {
        zero_f32<<<((int)((size_t)RALL * KPAD) + 255) / 256, 256, 0, stream>>>(bufB, RALL * KPAD);
        for (int g = 0; g < 3; ++g) {
            const float* feat = (l == 0) ? x[g] : (bufA + (size_t)g * NNODES * DH);
            int fs            = (l == 0) ? DIN  : DH;
            spmm_scatter<<<dim3(NEDGES, (D_l[l] + 127) / 128), 128, 0, stream>>>(
                feat, fs,
                bufB + (size_t)g * NNODES * KPAD, KPAD,
                rso + (size_t)g * NNODES,
                src[g], dst[g], D_l[l]);
        }
        scale_to_bf16<<<dim3((Ks_l[l] + 255) / 256, RALL), 256, 0, stream>>>(
            bufB, KPAD, rsi, hb, Ks_l[l]);
        wmma_gemm_relu<<<(n_tiles + 7) / 8, 256, 0, stream>>>(
            hb, wts[l], b[l], bufA, Ks_l[l], n_tiles);
    }

    // ---- global max pool + MLP head ----
    zero_f32<<<2, 256, 0, stream>>>(pool, DH);
    pool_max<<<dim3(2, (RALL + 255) / 256), 256, 0, stream>>>(bufA, pool, RALL);
    mlp_head<<<1, 128, 0, stream>>>(pool, fW1, fb1, fW2, fb2, fW3, fb3, (float*)d_out);
}
